// ContentStyleGraphNet_44461501448530
// MI455X (gfx1250) — matrix operand order repaired and checked by reading, not compiled, not run on hardware
//
#include <hip/hip_runtime.h>
#include <math.h>

typedef __attribute__((ext_vector_type(16))) _Float16 v16h;
typedef __attribute__((ext_vector_type(8)))  _Float16 v8h;
typedef __attribute__((ext_vector_type(8)))  float    v8f;

#define DEV __device__ __forceinline__

DEV float gelu_f(float v) { return 0.5f * v * (1.0f + erff(v * 0.70710678118654752f)); }
DEV float sigm_f(float v) { return 1.0f / (1.0f + __expf(-v)); }

// ---- WMMA fragment helpers (CDNA5 16x16x32 f16, wave32) -------------------
// A: 16x32 (MxK) from row-major f16 LDS. Per lane: two contiguous 8-half runs
// (k = hf*16.. and k = hf*16+8..) -> two ds_load_b128.
DEV v16h ld_afrag(const _Float16* base, int lda, int lane) {
  const int m = lane & 15, hf = lane >> 4;
  const _Float16* row = base + m * lda + hf * 16;
  const v8h lo = *(const v8h*)(row);
  const v8h hi = *(const v8h*)(row + 8);
  v16h a;
#pragma unroll
  for (int e = 0; e < 8; ++e) { a[e] = lo[e]; a[8 + e] = hi[e]; }
  return a;
}
// B: fragment-ready LDS layout: dst[(frag*32+lane)*16 + e] -> two ds_load_b128.
DEV v16h ld_bfrag(const _Float16* base, int frag, int lane) {
  const _Float16* p = base + (frag * 32 + lane) * 16;
  const v8h lo = *(const v8h*)(p);
  const v8h hi = *(const v8h*)(p + 8);
  v16h b;
#pragma unroll
  for (int e = 0; e < 8; ++e) { b[e] = lo[e]; b[8 + e] = hi[e]; }
  return b;
}
// Swizzle a row-major (N,K) f32 weight matrix W[n][k] (B = W^T, i.e. B[k][n])
// into fragment-ready f16 LDS. frag index = kt*NT + nt. K must be a power of 2.
DEV void swz_weight(_Float16* dst, const float* __restrict__ W, int N, int kbits, int tid) {
  const int K = 1 << kbits;
  const int NT = N >> 4;
  for (int i = tid; i < (N << kbits); i += 256) {
    const int n = i >> kbits, k = i & (K - 1);
    const int frag = (k >> 5) * NT + (n >> 4);
    const int lane = ((k >> 4) & 1) * 16 + (n & 15);
    dst[(frag * 32 + lane) * 16 + (k & 15)] = (_Float16)W[i];
  }
}
// K=160 variant for conv2 (N=64, K=32*5): frag = kt*4 + nt, kt in 0..4.
DEV void swz_weight160(_Float16* dst, const float* __restrict__ W, int tid) {
  for (int i = tid; i < 10240; i += 256) {
    const int n = i / 160, k = i - n * 160;
    const int frag = (k >> 5) * 4 + (n >> 4);
    const int lane = ((k >> 4) & 1) * 16 + (n & 15);
    dst[(frag * 32 + lane) * 16 + (k & 15)] = (_Float16)W[i];
  }
}
DEV v8f wmma32(v16h a, v16h b, v8f c) {
  return __builtin_amdgcn_wmma_f32_16x16x32_f16(false, a, false, b, (short)0, c, false, false);
}

// ===========================================================================
// 1) Fused per-(b,c) temporal CNN.
//    conv1+bn+gelu+pool4 on VALU (K=7, tiny); conv2 as im2col WMMA GEMM
//    (M=1024 positions, N=64, K=160 -> 5 chained WMMAs per tile) with
//    bn+gelu+maxpool4 fused in the C-fragment epilogue.
// ===========================================================================
__global__ void __launch_bounds__(256) conv_fused_kernel(
    const float* __restrict__ x,
    const float* __restrict__ c1w, const float* __restrict__ c1b,
    const float* __restrict__ bn1g, const float* __restrict__ bn1b,
    const float* __restrict__ bn1m, const float* __restrict__ bn1v,
    const float* __restrict__ c2w, const float* __restrict__ c2b,
    const float* __restrict__ bn2g, const float* __restrict__ bn2b,
    const float* __restrict__ bn2m, const float* __restrict__ bn2v,
    float* __restrict__ ht, float* __restrict__ h_avg)
{
  __shared__ __attribute__((aligned(16))) float xs[4096];
  __shared__ float h1[32 * 1028];
  __shared__ float w1s[224];
  __shared__ float a1c[32], b1c[32];
  __shared__ __attribute__((aligned(32))) _Float16 w2f[10240];   // frag-ready
  __shared__ __attribute__((aligned(32))) _Float16 sim[8 * 16 * 168]; // per-wave im2col
  __shared__ float a2c[64], b2c[64];
  __shared__ float sacc[64];
  const int bc = blockIdx.x;
  const int b = bc >> 4, c = bc & 15;
  const int tid = threadIdx.x;
  const int lane = tid & 31, wave = tid >> 5;

  {
    const float4* src = (const float4*)(x + (size_t)bc * 4096);
    for (int i = tid; i < 1024; i += 256) *(float4*)&xs[i * 4] = src[i];
  }
  for (int i = tid; i < 224; i += 256) w1s[i] = c1w[i];
  swz_weight160(w2f, c2w, tid);
  if (tid < 32) {
    float sc = bn1g[tid] * rsqrtf(bn1v[tid] + 1e-5f);
    a1c[tid] = sc;
    b1c[tid] = bn1b[tid] - bn1m[tid] * sc + c1b[tid] * sc;
  }
  if (tid < 64) {
    float sc = bn2g[tid] * rsqrtf(bn2v[tid] + 1e-5f);
    a2c[tid] = sc;
    b2c[tid] = bn2b[tid] - bn2m[tid] * sc + c2b[tid] * sc;
    sacc[tid] = 0.f;
  }
  __syncthreads();
  // conv1 (k=7, pad 3) + bn + gelu + maxpool4 -> h1 (32 x 1024)
  for (int idx = tid; idx < 32768; idx += 256) {
    const int oc = idx >> 10, tp = idx & 1023;
    const float A = a1c[oc], Bc = b1c[oc];
    float mx = -3.0e38f;
#pragma unroll
    for (int p = 0; p < 4; ++p) {
      const int pos = tp * 4 + p;
      float acc = 0.f;
#pragma unroll
      for (int j = 0; j < 7; ++j) {
        const int q = pos + j - 3;
        if (q >= 0 && q < 4096) acc += w1s[oc * 7 + j] * xs[q];
      }
      mx = fmaxf(mx, gelu_f(acc * A + Bc));
    }
    h1[oc * 1028 + tp] = mx;
  }
  __syncthreads();
  // conv2 via WMMA: each wave owns 8 M-tiles of 16 positions; wave-private
  // im2col staging (LDS ops are in-order within a wave -> no barrier needed).
  {
    _Float16* simw = &sim[wave * 16 * 168];
    const int nn = lane & 15, hf = lane >> 4;
    float psum[4] = {0.f, 0.f, 0.f, 0.f};
    for (int tl = 0; tl < 8; ++tl) {
      const int mt = wave * 8 + tl;               // 0..63 (position tile)
      for (int m = 0; m < 16; ++m) {
        const int p = mt * 16 + m;
        for (int k = lane; k < 160; k += 32) {
          const int ic = k / 5, j = k - ic * 5;
          const int q = p + j - 2;
          simw[m * 168 + k] =
              (q >= 0 && q < 1024) ? (_Float16)h1[ic * 1028 + q] : (_Float16)0.f;
        }
      }
      const v16h a0 = ld_afrag(simw, 168, lane);
      const v16h a1 = ld_afrag(simw + 32, 168, lane);
      const v16h a2 = ld_afrag(simw + 64, 168, lane);
      const v16h a3 = ld_afrag(simw + 96, 168, lane);
      const v16h a4 = ld_afrag(simw + 128, 168, lane);
#pragma unroll
      for (int nt = 0; nt < 4; ++nt) {
        v8f cc;
#pragma unroll
        for (int i = 0; i < 8; ++i) cc[i] = 0.f;
        cc = wmma32(a0, ld_bfrag(w2f, nt, lane), cc);
        cc = wmma32(a1, ld_bfrag(w2f, 4 + nt, lane), cc);
        cc = wmma32(a2, ld_bfrag(w2f, 8 + nt, lane), cc);
        cc = wmma32(a3, ld_bfrag(w2f, 12 + nt, lane), cc);
        cc = wmma32(a4, ld_bfrag(w2f, 16 + nt, lane), cc);
        const int col = nt * 16 + nn;
        const float A = a2c[col], Bc = b2c[col];
        float y[8];
#pragma unroll
        for (int i = 0; i < 8; ++i) y[i] = gelu_f(cc[i] * A + Bc);
        const float m0 = fmaxf(fmaxf(y[0], y[1]), fmaxf(y[2], y[3]));
        const float m1 = fmaxf(fmaxf(y[4], y[5]), fmaxf(y[6], y[7]));
        const int ta = mt * 4 + hf * 2;           // pooled t index (0..255)
        ht[(size_t)((b * 256 + ta) * 16 + c) * 64 + col] = m0;
        ht[(size_t)((b * 256 + ta + 1) * 16 + c) * 64 + col] = m1;
        psum[nt] += m0 + m1;
      }
    }
#pragma unroll
    for (int nt = 0; nt < 4; ++nt) atomicAdd(&sacc[nt * 16 + nn], psum[nt]);
  }
  __syncthreads();
  if (tid < 64) h_avg[(size_t)bc * 64 + tid] = sacc[tid] * (1.0f / 256.0f);
}

// ===========================================================================
// 2) center+normalize h_avg rows -> hn
// ===========================================================================
__global__ void __launch_bounds__(64) hn_kernel(const float* __restrict__ h_avg,
                                                float* __restrict__ hn)
{
  __shared__ float t0[64];
  const int bc = blockIdx.x, d = threadIdx.x;
  const float v = h_avg[(size_t)bc * 64 + d];
  t0[d] = v; __syncthreads();
  for (int s = 32; s > 0; s >>= 1) { if (d < s) t0[d] += t0[d + s]; __syncthreads(); }
  const float mean = t0[0] * (1.f / 64.f);
  __syncthreads();
  const float hc = v - mean;
  t0[d] = hc * hc; __syncthreads();
  for (int s = 32; s > 0; s >>= 1) { if (d < s) t0[d] += t0[d + s]; __syncthreads(); }
  const float nrm = sqrtf(t0[0]) + 1e-8f;
  hn[(size_t)bc * 64 + d] = hc / nrm;
}

// ===========================================================================
// 3) In-LDS 4096-pt radix-2 FFT per (b,c); writes normalized |rfft| (2049)
// ===========================================================================
__global__ void __launch_bounds__(256) fft_kernel(const float* __restrict__ x,
                                                  float* __restrict__ Xn)
{
  __shared__ float re[4096], im[4096];
  __shared__ float red[256];
  const int bc = blockIdx.x, tid = threadIdx.x;
  for (int i = tid; i < 4096; i += 256) {
    const int r = (int)(__brev((unsigned)i) >> 20);   // 12-bit reversal
    re[i] = x[(size_t)bc * 4096 + r];
    im[i] = 0.f;
  }
  __syncthreads();
  for (int s = 0; s < 12; ++s) {
    const int half = 1 << s, m = half << 1;
    for (int bf = tid; bf < 2048; bf += 256) {
      const int blk = bf >> s, pos = bf & (half - 1);
      const int i0 = blk * m + pos, i1 = i0 + half;
      const float ang = -6.283185307179586f * (float)pos / (float)m;
      const float wr = cosf(ang), wi = sinf(ang);
      const float xr = re[i1], xi = im[i1];
      const float tr = wr * xr - wi * xi, ti = wr * xi + wi * xr;
      const float ur = re[i0], ui = im[i0];
      re[i0] = ur + tr; im[i0] = ui + ti;
      re[i1] = ur - tr; im[i1] = ui - ti;
    }
    __syncthreads();
  }
  float ss = 0.f;
  for (int f = tid; f < 2049; f += 256) {
    const float m2 = re[f] * re[f] + im[f] * im[f];
    re[f] = sqrtf(m2);
    ss += m2;
  }
  red[tid] = ss; __syncthreads();
  for (int s = 128; s > 0; s >>= 1) { if (tid < s) red[tid] += red[tid + s]; __syncthreads(); }
  const float nrm = sqrtf(red[0]) + 1e-8f;
  for (int f = tid; f < 2049; f += 256) Xn[(size_t)bc * 2052 + f] = re[f] / nrm;
}

// ===========================================================================
// 4) adjacency: softmax(logw)-blend of feat_corr, spec_coh, symmetric prior
// ===========================================================================
__global__ void __launch_bounds__(256) adj_kernel(
    const float* __restrict__ hn, const float* __restrict__ Xn,
    const float* __restrict__ prior, const float* __restrict__ lw,
    float* __restrict__ adjws, float* __restrict__ adjout)
{
  const int b = blockIdx.x, tid = threadIdx.x;
  const int c = tid >> 4, e = tid & 15;
  const float w0 = lw[0], w1 = lw[1], w2 = lw[2];
  const float mw = fmaxf(w0, fmaxf(w1, w2));
  const float e0 = __expf(w0 - mw), e1 = __expf(w1 - mw), e2 = __expf(w2 - mw);
  const float inv = 1.f / (e0 + e1 + e2);
  const float* hc = hn + (size_t)(b * 16 + c) * 64;
  const float* he = hn + (size_t)(b * 16 + e) * 64;
  float fc = 0.f;
#pragma unroll
  for (int dd = 0; dd < 64; ++dd) fc += hc[dd] * he[dd];
  const float* xc = Xn + (size_t)(b * 16 + c) * 2052;
  const float* xe = Xn + (size_t)(b * 16 + e) * 2052;
  float sc = 0.f;
  for (int f = 0; f < 2049; ++f) sc += xc[f] * xe[f];
  const float pv = prior[c * 16 + e] + prior[e * 16 + c];
  const float a = (e0 * fc + e1 * sc + e2 * pv) * inv;
  adjws[b * 256 + tid] = a;
  adjout[b * 256 + tid] = a;
}

// ===========================================================================
// 5) Fused GAT transformer layer (8 graphs x 16 nodes per WG, all in LDS)
// ===========================================================================
#define SXS 68
#define SAS 72
#define SMS 200

__global__ void __launch_bounds__(256) gat_layer_kernel(
    float* __restrict__ ht, const float* __restrict__ adj,
    const float* __restrict__ n1g, const float* __restrict__ n1b,
    const float* __restrict__ qkvw, const float* __restrict__ qkvb,
    const float* __restrict__ outw, const float* __restrict__ outb,
    const float* __restrict__ n2g, const float* __restrict__ n2b,
    const float* __restrict__ f1w, const float* __restrict__ f1b,
    const float* __restrict__ f2w, const float* __restrict__ f2b)
{
  __shared__ __attribute__((aligned(16))) float    sx[128 * SXS];
  __shared__ __attribute__((aligned(32))) _Float16 sa[128 * SAS];
  __shared__ __attribute__((aligned(32))) _Float16 sm[128 * SMS];
  __shared__ __attribute__((aligned(32))) _Float16 swb[64 * 192];
  __shared__ float sbias[192];
  __shared__ float sadj[256];
  __shared__ float slng[64], slnb[64];

  const int tid = threadIdx.x;
  const int lane = tid & 31, wave = tid >> 5;
  const int n0 = blockIdx.x * 8;
  const int bidx = n0 >> 8;                 // Tp=256 divides the tile
  const size_t gbase = (size_t)n0 * 16 * 64;

  // phase0: loads (residual tile as float4, weights swizzled to frag layout)
  {
    const float4* src = (const float4*)(ht + gbase);
    for (int i = tid; i < 2048; i += 256) {
      const int r = i >> 4, q = i & 15;
      *(float4*)&sx[r * SXS + q * 4] = src[i];
    }
  }
  sadj[tid] = adj[bidx * 256 + tid];
  if (tid < 64) { slng[tid] = n1g[tid]; slnb[tid] = n1b[tid]; }
  swz_weight(swb, qkvw, 192, 6, tid);
  if (tid < 192) sbias[tid] = qkvb[tid];
  __syncthreads();

  // phase1: LN1 -> sa (f16, vector stores)
  if (tid < 128) {
    const float* px = &sx[tid * SXS];
    float m = 0.f;
#pragma unroll
    for (int dd = 0; dd < 64; ++dd) m += px[dd];
    m *= (1.f / 64.f);
    float v = 0.f;
#pragma unroll
    for (int dd = 0; dd < 64; ++dd) { const float t = px[dd] - m; v += t * t; }
    const float rs = rsqrtf(v * (1.f / 64.f) + 1e-5f);
    _Float16* pa = &sa[tid * SAS];
#pragma unroll
    for (int vv = 0; vv < 8; ++vv) {
      v8h h;
#pragma unroll
      for (int e = 0; e < 8; ++e) {
        const int dd = vv * 8 + e;
        h[e] = (_Float16)((px[dd] - m) * rs * slng[dd] + slnb[dd]);
      }
      *(v8h*)(pa + vv * 8) = h;
    }
  }
  __syncthreads();

  // phase2: qkv GEMM (128x64 @ 64x192) via WMMA  (NT=12)
  {
    const int mt = wave;
    const v16h a0 = ld_afrag(&sa[mt * 16 * SAS], SAS, lane);
    const v16h a1 = ld_afrag(&sa[mt * 16 * SAS + 32], SAS, lane);
    const int nn = lane & 15, hf = lane >> 4;
    for (int nt = 0; nt < 12; ++nt) {
      const v16h b0 = ld_bfrag(swb, nt, lane);
      const v16h b1 = ld_bfrag(swb, 12 + nt, lane);
      v8f cc;
      const float bv = sbias[nt * 16 + nn];
#pragma unroll
      for (int i = 0; i < 8; ++i) cc[i] = bv;
      cc = wmma32(a0, b0, cc);
      cc = wmma32(a1, b1, cc);
#pragma unroll
      for (int i = 0; i < 8; ++i) {
        const int row = mt * 16 + i + hf * 8;
        sm[row * SMS + nt * 16 + nn] = (_Float16)cc[i];
      }
    }
  }
  __syncthreads();

  // phase3: 16x16 multi-head attention (vectorized LDS) + preload Wout/LN2
  swz_weight(swb, outw, 64, 6, tid);
  if (tid < 64) { sbias[tid] = outb[tid]; slng[tid] = n2g[tid]; slnb[tid] = n2b[tid]; }
  {
    const int iq = lane & 15, hf = lane >> 4;
    for (int it = 0; it < 2; ++it) {
      const int pr = wave * 4 + it * 2 + hf;
      const int g = pr >> 2, hh = pr & 3;
      const _Float16* base = &sm[(g * 16) * SMS + hh * 16];
      const v8h q0 = *(const v8h*)(base + iq * SMS);
      const v8h q1 = *(const v8h*)(base + iq * SMS + 8);
      float s[16];
      float mx = -3.0e38f;
      for (int j = 0; j < 16; ++j) {
        const _Float16* kr = base + j * SMS + 64;
        const v8h k0 = *(const v8h*)kr;
        const v8h k1 = *(const v8h*)(kr + 8);
        float acc = 0.f;
#pragma unroll
        for (int e = 0; e < 8; ++e)
          acc += (float)q0[e] * (float)k0[e] + (float)q1[e] * (float)k1[e];
        s[j] = acc * 0.25f + sadj[iq * 16 + j];
        mx = fmaxf(mx, s[j]);
      }
      float ssum = 0.f;
#pragma unroll
      for (int j = 0; j < 16; ++j) { s[j] = __expf(s[j] - mx); ssum += s[j]; }
      const float inv = 1.f / ssum;
      float o[16];
#pragma unroll
      for (int e = 0; e < 16; ++e) o[e] = 0.f;
      for (int j = 0; j < 16; ++j) {
        const _Float16* vr = base + j * SMS + 128;
        const v8h v0 = *(const v8h*)vr;
        const v8h v1 = *(const v8h*)(vr + 8);
        const float pj = s[j];
#pragma unroll
        for (int e = 0; e < 8; ++e) { o[e] += pj * (float)v0[e]; o[8 + e] += pj * (float)v1[e]; }
      }
      v8h r0, r1;
#pragma unroll
      for (int e = 0; e < 8; ++e) {
        r0[e] = (_Float16)(o[e] * inv);
        r1[e] = (_Float16)(o[8 + e] * inv);
      }
      _Float16* dst = &sa[(g * 16 + iq) * SAS + hh * 16];
      *(v8h*)dst = r0;
      *(v8h*)(dst + 8) = r1;
    }
  }
  __syncthreads();

  // phase4: out-proj GEMM with residual (C = sx + bias)  (NT=4)
  {
    const int mt = wave;
    const v16h a0 = ld_afrag(&sa[mt * 16 * SAS], SAS, lane);
    const v16h a1 = ld_afrag(&sa[mt * 16 * SAS + 32], SAS, lane);
    const int nn = lane & 15, hf = lane >> 4;
    for (int nt = 0; nt < 4; ++nt) {
      const v16h b0 = ld_bfrag(swb, nt, lane);
      const v16h b1 = ld_bfrag(swb, 4 + nt, lane);
      v8f cc;
      const float bv = sbias[nt * 16 + nn];
#pragma unroll
      for (int i = 0; i < 8; ++i) {
        const int row = mt * 16 + i + hf * 8;
        cc[i] = sx[row * SXS + nt * 16 + nn] + bv;
      }
      cc = wmma32(a0, b0, cc);
      cc = wmma32(a1, b1, cc);
#pragma unroll
      for (int i = 0; i < 8; ++i) {
        const int row = mt * 16 + i + hf * 8;
        sx[row * SXS + nt * 16 + nn] = cc[i];
      }
    }
  }
  __syncthreads();

  // phase5: LN2 -> sa ; preload W1/f1b
  swz_weight(swb, f1w, 128, 6, tid);
  if (tid < 128) sbias[tid] = f1b[tid];
  if (tid < 128) {
    const float* px = &sx[tid * SXS];
    float m = 0.f;
#pragma unroll
    for (int dd = 0; dd < 64; ++dd) m += px[dd];
    m *= (1.f / 64.f);
    float v = 0.f;
#pragma unroll
    for (int dd = 0; dd < 64; ++dd) { const float t = px[dd] - m; v += t * t; }
    const float rs = rsqrtf(v * (1.f / 64.f) + 1e-5f);
    _Float16* pa = &sa[tid * SAS];
#pragma unroll
    for (int vv = 0; vv < 8; ++vv) {
      v8h h;
#pragma unroll
      for (int e = 0; e < 8; ++e) {
        const int dd = vv * 8 + e;
        h[e] = (_Float16)((px[dd] - m) * rs * slng[dd] + slnb[dd]);
      }
      *(v8h*)(pa + vv * 8) = h;
    }
  }
  __syncthreads();

  // phase6: FFN up (128x64 @ 64x128) + exact GELU -> sm  (NT=8)
  {
    const int mt = wave;
    const v16h a0 = ld_afrag(&sa[mt * 16 * SAS], SAS, lane);
    const v16h a1 = ld_afrag(&sa[mt * 16 * SAS + 32], SAS, lane);
    const int nn = lane & 15, hf = lane >> 4;
    for (int nt = 0; nt < 8; ++nt) {
      const v16h b0 = ld_bfrag(swb, nt, lane);
      const v16h b1 = ld_bfrag(swb, 8 + nt, lane);
      v8f cc;
      const float bv = sbias[nt * 16 + nn];
#pragma unroll
      for (int i = 0; i < 8; ++i) cc[i] = bv;
      cc = wmma32(a0, b0, cc);
      cc = wmma32(a1, b1, cc);
#pragma unroll
      for (int i = 0; i < 8; ++i) {
        const int row = mt * 16 + i + hf * 8;
        sm[row * SMS + nt * 16 + nn] = (_Float16)gelu_f(cc[i]);
      }
    }
  }
  __syncthreads();

  // phase7: preload W2/f2b (K=128 -> NT=4, kt=0..3)
  swz_weight(swb, f2w, 64, 7, tid);
  if (tid < 64) sbias[tid] = f2b[tid];
  __syncthreads();

  // phase8: FFN down (K=128, 4 chained WMMAs) + residual -> global ht
  {
    const int mt = wave;
    const v16h a0 = ld_afrag(&sm[mt * 16 * SMS], SMS, lane);
    const v16h a1 = ld_afrag(&sm[mt * 16 * SMS + 32], SMS, lane);
    const v16h a2 = ld_afrag(&sm[mt * 16 * SMS + 64], SMS, lane);
    const v16h a3 = ld_afrag(&sm[mt * 16 * SMS + 96], SMS, lane);
    const int nn = lane & 15, hf = lane >> 4;
    for (int nt = 0; nt < 4; ++nt) {
      const v16h b0 = ld_bfrag(swb, nt, lane);
      const v16h b1 = ld_bfrag(swb, 4 + nt, lane);
      const v16h b2 = ld_bfrag(swb, 8 + nt, lane);
      const v16h b3 = ld_bfrag(swb, 12 + nt, lane);
      v8f cc;
      const float bv = sbias[nt * 16 + nn];
#pragma unroll
      for (int i = 0; i < 8; ++i) {
        const int row = mt * 16 + i + hf * 8;
        cc[i] = sx[row * SXS + nt * 16 + nn] + bv;
      }
      cc = wmma32(a0, b0, cc);
      cc = wmma32(a1, b1, cc);
      cc = wmma32(a2, b2, cc);
      cc = wmma32(a3, b3, cc);
#pragma unroll
      for (int i = 0; i < 8; ++i) {
        const int row = mt * 16 + i + hf * 8;
        ht[gbase + (size_t)row * 64 + nt * 16 + nn] = cc[i];
      }
    }
  }
}

// ===========================================================================
// 6) final LayerNorm + permute (B,Tp,C,D) -> hs16 (B*C, Tp, D) f16
// ===========================================================================
__global__ void __launch_bounds__(256) final_ln_kernel(
    const float* __restrict__ ht, const float* __restrict__ gg,
    const float* __restrict__ gb, _Float16* __restrict__ hs16)
{
  __shared__ float sg[64], sb2[64];
  if (threadIdx.x < 64) { sg[threadIdx.x] = gg[threadIdx.x]; sb2[threadIdx.x] = gb[threadIdx.x]; }
  __syncthreads();
  const int row = blockIdx.x * 256 + threadIdx.x;   // 0..131071
  const int n = row >> 4, c = row & 15;
  const int b = n >> 8, t = n & 255;
  const float* px = ht + (size_t)row * 64;
  float m = 0.f;
#pragma unroll
  for (int dd = 0; dd < 64; ++dd) m += px[dd];
  m *= (1.f / 64.f);
  float v = 0.f;
#pragma unroll
  for (int dd = 0; dd < 64; ++dd) { const float tt = px[dd] - m; v += tt * tt; }
  const float rs = rsqrtf(v * (1.f / 64.f) + 1e-5f);
  _Float16* po = hs16 + (size_t)((b * 16 + c) * 256 + t) * 64;
#pragma unroll
  for (int vv = 0; vv < 8; ++vv) {
    v8h h;
#pragma unroll
    for (int e = 0; e < 8; ++e) {
      const int dd = vv * 8 + e;
      h[e] = (_Float16)((px[dd] - m) * rs * sg[dd] + sb2[dd]);
    }
    *(v8h*)(po + vv * 8) = h;
  }
}

// ===========================================================================
// 7) Bidirectional GRU: 32 WGs (2 dirs x 16 chunks of 32 seqs).
//    One (mt,ng) unit per wave -> 12-WMMA critical path per timestep.
// ===========================================================================
__global__ void __launch_bounds__(256) gru_kernel(
    const _Float16* __restrict__ hs16,
    const float* __restrict__ wih, const float* __restrict__ whh,
    const float* __restrict__ bih, const float* __restrict__ bhh,
    float* __restrict__ h_out)
{
  __shared__ float    sh[32 * 68];
  __shared__ __attribute__((aligned(32))) _Float16 sha[2][32 * 72];
  __shared__ __attribute__((aligned(32))) _Float16 sxa[32 * 72];
  __shared__ __attribute__((aligned(32))) _Float16 swi[64 * 192];
  __shared__ __attribute__((aligned(32))) _Float16 swh[64 * 192];
  __shared__ float sbi[192], sbh[192];

  const int tid = threadIdx.x, lane = tid & 31, wave = tid >> 5;
  const int dir = blockIdx.x >> 4, chunk = blockIdx.x & 15;
  const int s0 = chunk * 32;

  swz_weight(swi, wih + (size_t)dir * 12288, 192, 6, tid);
  swz_weight(swh, whh + (size_t)dir * 12288, 192, 6, tid);
  if (tid < 192) { sbi[tid] = bih[dir * 192 + tid]; sbh[tid] = bhh[dir * 192 + tid]; }
  for (int i = tid; i < 32 * 68; i += 256) sh[i] = 0.f;
  {
    _Float16* pz = &sha[0][0];
    for (int i = tid; i < 2 * 32 * 72; i += 256) pz[i] = (_Float16)0.f;
  }
  __syncthreads();

  const int mt = wave & 1;          // M-tile (rows mt*16..+15)
  const int ng = wave >> 1;         // N column group (cols ng*16..+15)
  const int nn = lane & 15, hf = lane >> 4;
  const int col = ng * 16 + nn;
  for (int step = 0; step < 256; ++step) {
    const int tt = dir ? (255 - step) : step;
    // x_t tile (32 rows x 64), one v8h per thread
    {
      const int r = tid >> 3, q = tid & 7;
      *(v8h*)&sxa[r * 72 + q * 8] =
          *(const v8h*)(hs16 + ((size_t)(s0 + r) * 256 + (size_t)tt) * 64 + q * 8);
    }
    // prefetch next timestep's tile into cache (global_prefetch_b8)
    if (step + 1 < 256) {
      const int tn = dir ? (254 - step) : (step + 1);
      const _Float16* np =
          hs16 + ((size_t)(s0 + (tid >> 3)) * 256 + (size_t)tn) * 64 + (tid & 7) * 8;
      __builtin_prefetch(np, 0, 1);
    }
    __syncthreads();
    const _Float16* hA = &sha[step & 1][0];
    _Float16* hN = &sha[(step & 1) ^ 1][0];
    const v16h ax0 = ld_afrag(&sxa[mt * 16 * 72], 72, lane);
    const v16h ax1 = ld_afrag(&sxa[mt * 16 * 72 + 32], 72, lane);
    const v16h ah0 = ld_afrag(&hA[mt * 16 * 72], 72, lane);
    const v16h ah1 = ld_afrag(&hA[mt * 16 * 72 + 32], 72, lane);
    v8f gi_r, gi_z, gi_n, gh_r, gh_z, gh_n;
#pragma unroll
    for (int i = 0; i < 8; ++i) {
      gi_r[i] = sbi[col]; gi_z[i] = sbi[64 + col]; gi_n[i] = sbi[128 + col];
      gh_r[i] = sbh[col]; gh_z[i] = sbh[64 + col]; gh_n[i] = sbh[128 + col];
    }
    {
      const v16h b0 = ld_bfrag(swi, ng, lane);
      const v16h b1 = ld_bfrag(swi, 12 + ng, lane);
      gi_r = wmma32(ax0, b0, gi_r); gi_r = wmma32(ax1, b1, gi_r);
    }
    {
      const v16h b0 = ld_bfrag(swi, 4 + ng, lane);
      const v16h b1 = ld_bfrag(swi, 16 + ng, lane);
      gi_z = wmma32(ax0, b0, gi_z); gi_z = wmma32(ax1, b1, gi_z);
    }
    {
      const v16h b0 = ld_bfrag(swi, 8 + ng, lane);
      const v16h b1 = ld_bfrag(swi, 20 + ng, lane);
      gi_n = wmma32(ax0, b0, gi_n); gi_n = wmma32(ax1, b1, gi_n);
    }
    {
      const v16h b0 = ld_bfrag(swh, ng, lane);
      const v16h b1 = ld_bfrag(swh, 12 + ng, lane);
      gh_r = wmma32(ah0, b0, gh_r); gh_r = wmma32(ah1, b1, gh_r);
    }
    {
      const v16h b0 = ld_bfrag(swh, 4 + ng, lane);
      const v16h b1 = ld_bfrag(swh, 16 + ng, lane);
      gh_z = wmma32(ah0, b0, gh_z); gh_z = wmma32(ah1, b1, gh_z);
    }
    {
      const v16h b0 = ld_bfrag(swh, 8 + ng, lane);
      const v16h b1 = ld_bfrag(swh, 20 + ng, lane);
      gh_n = wmma32(ah0, b0, gh_n); gh_n = wmma32(ah1, b1, gh_n);
    }
#pragma unroll
    for (int i = 0; i < 8; ++i) {
      const int row = mt * 16 + i + hf * 8;
      const float hp = sh[row * 68 + col];
      const float rg = sigm_f(gi_r[i] + gh_r[i]);
      const float zg = sigm_f(gi_z[i] + gh_z[i]);
      const float ngv = tanhf(gi_n[i] + rg * gh_n[i]);
      const float hv = (1.f - zg) * ngv + zg * hp;
      sh[row * 68 + col] = hv;
      hN[row * 72 + col] = (_Float16)hv;
    }
    __syncthreads();
  }
  for (int i = tid; i < 2048; i += 256) {
    const int r = i >> 6, d2 = i & 63;
    h_out[(size_t)(s0 + r) * 128 + (size_t)dir * 64 + d2] = sh[r * 68 + d2];
  }
}

// ===========================================================================
// 8) gated channel readout
// ===========================================================================
__global__ void __launch_bounds__(128) readout_kernel(
    const float* __restrict__ h_out, const float* __restrict__ gw,
    const float* __restrict__ gb, float* __restrict__ out)
{
  __shared__ float gsh[16];
  const int b = blockIdx.x, tid = threadIdx.x;
  if (tid < 16) {
    const float* hp = h_out + (size_t)(b * 16 + tid) * 128;
    float acc = gb[0];
    for (int dd = 0; dd < 128; ++dd) acc += hp[dd] * gw[dd];
    const float gv = 1.f / (1.f + __expf(-acc));
    gsh[tid] = gv;
    out[12288 + b * 16 + tid] = gv;
  }
  __syncthreads();
  float gs = 0.f;
#pragma unroll
  for (int cc = 0; cc < 16; ++cc) gs += gsh[cc];
  float acc = 0.f;
  for (int cc = 0; cc < 16; ++cc) acc += h_out[(size_t)(b * 16 + cc) * 128 + tid] * gsh[cc];
  out[b * 128 + tid] = acc / (gs + 1e-8f);
}

// ===========================================================================
extern "C" void kernel_launch(void* const* d_in, const int* in_sizes, int n_in,
                              void* d_out, int out_size, void* d_ws, size_t ws_size,
                              hipStream_t stream)
{
  (void)in_sizes; (void)n_in; (void)out_size; (void)ws_size;
  const float* x       = (const float*)d_in[0];
  const float* c1w     = (const float*)d_in[1];
  const float* c1b     = (const float*)d_in[2];
  const float* bn1g    = (const float*)d_in[3];
  const float* bn1b    = (const float*)d_in[4];
  const float* bn1m    = (const float*)d_in[5];
  const float* bn1v    = (const float*)d_in[6];
  const float* c2w     = (const float*)d_in[7];
  const float* c2b     = (const float*)d_in[8];
  const float* bn2g    = (const float*)d_in[9];
  const float* bn2b    = (const float*)d_in[10];
  const float* bn2m    = (const float*)d_in[11];
  const float* bn2v    = (const float*)d_in[12];
  const float* prior   = (const float*)d_in[13];
  const float* logw    = (const float*)d_in[14];
  const float* n1g     = (const float*)d_in[15];
  const float* n1b     = (const float*)d_in[16];
  const float* qkvw    = (const float*)d_in[17];
  const float* qkvb    = (const float*)d_in[18];
  const float* outw    = (const float*)d_in[19];
  const float* outb    = (const float*)d_in[20];
  const float* n2g     = (const float*)d_in[21];
  const float* n2b     = (const float*)d_in[22];
  const float* f1w     = (const float*)d_in[23];
  const float* f1b     = (const float*)d_in[24];
  const float* f2w     = (const float*)d_in[25];
  const float* f2b     = (const float*)d_in[26];
  const float* gatng   = (const float*)d_in[27];
  const float* gatnb   = (const float*)d_in[28];
  const float* gruwih  = (const float*)d_in[29];
  const float* gruwhh  = (const float*)d_in[30];
  const float* grubih  = (const float*)d_in[31];
  const float* grubhh  = (const float*)d_in[32];
  const float* gatew   = (const float*)d_in[33];
  const float* gateb   = (const float*)d_in[34];

  char* w = (char*)d_ws;
  float*    ht    = (float*)w;     w += 33554432;   // 8192*16*64 f32
  _Float16* hs16  = (_Float16*)w;  w += 16777216;   // 512*256*64 f16
  float*    Xn    = (float*)w;     w += 4202496;    // 512*2052 f32
  float*    havg  = (float*)w;     w += 131072;     // 512*64 f32
  float*    hnb   = (float*)w;     w += 131072;     // 512*64 f32
  float*    hout  = (float*)w;     w += 262144;     // 512*128 f32
  float*    adjws = (float*)w;     w += 32768;      // 32*256 f32
  float* outp = (float*)d_out;

  conv_fused_kernel<<<512, 256, 0, stream>>>(x, c1w, c1b, bn1g, bn1b, bn1m, bn1v,
                                             c2w, c2b, bn2g, bn2b, bn2m, bn2v, ht, havg);
  hn_kernel<<<512, 64, 0, stream>>>(havg, hnb);
  fft_kernel<<<512, 256, 0, stream>>>(x, Xn);
  adj_kernel<<<32, 256, 0, stream>>>(hnb, Xn, prior, logw, adjws, outp + 4096);
  for (int l = 0; l < 2; ++l) {
    gat_layer_kernel<<<1024, 256, 0, stream>>>(ht, adjws,
        n1g + l * 64, n1b + l * 64, qkvw + l * 12288, qkvb + l * 192,
        outw + l * 4096, outb + l * 64, n2g + l * 64, n2b + l * 64,
        f1w + l * 8192, f1b + l * 128, f2w + l * 8192, f2b + l * 64);
  }
  final_ln_kernel<<<512, 256, 0, stream>>>(ht, gatng, gatnb, hs16);
  gru_kernel<<<32, 256, 0, stream>>>(hs16, gruwih, gruwhh, grubih, grubhh, hout);
  readout_kernel<<<32, 128, 0, stream>>>(hout, gatew, gateb, outp);
}
